// Attention_10179072491926
// MI455X (gfx1250) — matrix operand order repaired and checked by reading, not compiled, not run on hardware
//
#include <hip/hip_runtime.h>
#include <hip/hip_bf16.h>

// ---------------------------------------------------------------------------
// Problem constants (from the reference)
// ---------------------------------------------------------------------------
#define BATCH   2
#define SEQ     2048
#define DIM     1024
#define HEADS   16
#define DHEAD   64
#define INNER   1024          // HEADS * DHEAD
#define NQKV    3072          // 3 * INNER
#define ATTN_SCALE 0.03125f   // DIM^-0.5 = 1/32 (exact in bf16/f32)

typedef __bf16 bf16_t;
typedef __attribute__((ext_vector_type(16))) __bf16        v16bf;
typedef __attribute__((ext_vector_type(8)))  __bf16        v8bf;
typedef __attribute__((ext_vector_type(8)))  float         v8f;
typedef __attribute__((ext_vector_type(4)))  unsigned int  v4u;
typedef __attribute__((ext_vector_type(8)))  int           v8i;
typedef __attribute__((ext_vector_type(4)))  int           v4i;

// ---------------------------------------------------------------------------
// WMMA fragment helpers (CDNA5 wave32, V_WMMA_F32_16X16X32_BF16)
//
// A (16x32, MxK): lane l -> M=l%16, holds K = h*16+0..15 (contiguous,
// matching the B fragment), where h = lane>=16.
// ---------------------------------------------------------------------------
static __device__ inline v16bf load_a_frag(const bf16_t* base, int stride, int lane) {
    const int r = lane & 15;
    const int h = (lane >> 4) & 1;
    const bf16_t* row = base + r * stride + h * 16;
    union { v16bf v; v8bf p[2]; } u;
    u.p[0] = *reinterpret_cast<const v8bf*>(row);        // K = h*16 + 0..7
    u.p[1] = *reinterpret_cast<const v8bf*>(row + 8);    // K = h*16 + 8..15
    return u.v;
}

// B (32x16, KxN): lanes 0-15 hold K=0..15 (2 per VGPR), lanes 16-31 K=16..31.
// Source must be [N][K] layout (one contraction-contiguous row per lane).
static __device__ inline v16bf load_b_frag(const bf16_t* base, int stride, int lane) {
    const bf16_t* row = base + (lane & 15) * stride + ((lane >> 4) & 1) * 16;
    return *reinterpret_cast<const v16bf*>(row);         // 32 contiguous bytes
}

static __device__ inline v8f wmma_bf16(v16bf a, v16bf b, v8f c) {
    // (neg_a, A, neg_b, B, c_mod, C, reuse_a, reuse_b)
    return __builtin_amdgcn_wmma_f32_16x16x32_bf16(false, a, false, b,
                                                   (short)0, c, false, false);
}

// ---------------------------------------------------------------------------
// TDM: async-load a fully contiguous region (here a 64x64 bf16 tile = 8192B)
// from global into LDS. Descriptor per ISA ch.8:
//   group0: [count=1][lds_addr][global_addr lo][global_addr hi | type=2]
//   group1: data_size=4B, 1 row of `nwords` elements, strides benign.
// This toolchain exposes the 6-arg builtin:
//   (uint32x4 g0, int32x8 g1, int32x4 g2, int32x4 g3, int32x8 g4, i32 cpol)
// Caller must issue from a single wave and follow with s_wait_tensorcnt(0).
// ---------------------------------------------------------------------------
static __device__ inline void tdm_load_contig(void* lds_dst, const void* gsrc,
                                              unsigned int nwords /* 4B units */) {
    const unsigned long long ga = (unsigned long long)(uintptr_t)gsrc;
    const unsigned int lds_off  = (unsigned int)(uintptr_t)lds_dst; // addr[31:0] = LDS offset

    v4u g0;
    g0[0] = 1u;                                  // count=1, user descriptor
    g0[1] = lds_off;                             // lds_addr (bytes)
    g0[2] = (unsigned int)ga;                    // global_addr[31:0]
    g0[3] = ((unsigned int)(ga >> 32) & 0x01FFFFFFu) | 0x80000000u; // [56:32] | type=2

    v8i g1;
    g1[0] = 0x00020000;                          // workgroup_mask=0, data_size=2 (4B)
    g1[1] = (int)(nwords << 16);                 // tensor_dim0[15:0]  (w1[31:16])
    g1[2] = 0x00010000 | (int)(nwords >> 16);    // tensor_dim0[31:16] | tensor_dim1=1
    g1[3] = (int)(nwords << 16);                 // tile_dim0 = nwords (w3[31:16])
    g1[4] = 1;                                   // tile_dim1 = 1, tile_dim2 = 0
    g1[5] = (int)nwords;                         // tensor_dim0_stride[31:0]
    g1[6] = (int)(nwords << 16);                 // dim0_stride[47:32]=0 | dim1_stride[15:0]
    g1[7] = 0;                                   // tensor_dim1_stride[47:16] = 0

    v4i gz4 = {0, 0, 0, 0};                      // groups 2/3 unused (tile dims 0)
    v8i gz8 = {0, 0, 0, 0, 0, 0, 0, 0};
    __builtin_amdgcn_tensor_load_to_lds(g0, g1, gz4, gz4, gz8, 0);
}

// ---------------------------------------------------------------------------
// Kernel 0: convert f32 operands to bf16 staging buffers (read-many operands)
// ---------------------------------------------------------------------------
__global__ void cvt_bf16_kernel(const float* __restrict__ x,   bf16_t* __restrict__ xb,
                                const float* __restrict__ wq,  bf16_t* __restrict__ wqb,
                                const float* __restrict__ wo,  bf16_t* __restrict__ wob) {
    const size_t i  = (size_t)blockIdx.x * blockDim.x + threadIdx.x;
    const size_t gs = (size_t)gridDim.x * blockDim.x;
    for (size_t t = i; t < (size_t)BATCH * SEQ * DIM; t += gs) xb[t]  = (bf16_t)x[t];
    for (size_t t = i; t < (size_t)DIM * NQKV;        t += gs) wqb[t] = (bf16_t)wq[t];
    for (size_t t = i; t < (size_t)DIM * DIM;         t += gs) wob[t] = (bf16_t)wo[t];
}

// ---------------------------------------------------------------------------
// Kernel 1: QKV projection GEMM  C[4096,3072] = Xb[4096,1024] @ Wb[1024,3072]
// 128 threads (4 waves). Workgroup tile 128(M)x64(N); each wave owns a 32x64
// strip (two A fragments share every B fragment -> 8 WMMA per 4 LDS B-frags).
// Epilogue scatters to Q/K/V in [b*H + h][n][d] bf16 layout, scale on Q.
// ---------------------------------------------------------------------------
__global__ void __launch_bounds__(128)
qkv_gemm_kernel(const bf16_t* __restrict__ Xb, const bf16_t* __restrict__ Wb,
                bf16_t* __restrict__ Q, bf16_t* __restrict__ K, bf16_t* __restrict__ V) {
    __shared__ __align__(32) bf16_t Bt[64][64];   // W tile, [N][K]

    const int tid  = threadIdx.x;
    const int lane = tid & 31;
    const int wave = tid >> 5;
    const int n0   = blockIdx.x * 64;
    const int m0   = blockIdx.y * 128;

    v8f acc[2][4] = {};

    for (int k0 = 0; k0 < DIM; k0 += 64) {
        // Stage W[k0..k0+64][n0..n0+64] transposed into Bt[n][k]
        {
            const int k  = tid >> 1;
            const int c0 = (tid & 1) * 32;
            const bf16_t* src = Wb + (size_t)(k0 + k) * NQKV + n0 + c0;
#pragma unroll
            for (int c = 0; c < 32; ++c) Bt[c0 + c][k] = src[c];
        }
        __syncthreads();
#pragma unroll
        for (int kf = 0; kf < 2; ++kf) {
            v16bf a0 = load_a_frag(Xb + (size_t)(m0 + wave * 32) * DIM + k0 + kf * 32,
                                   DIM, lane);
            v16bf a1 = load_a_frag(Xb + (size_t)(m0 + wave * 32 + 16) * DIM + k0 + kf * 32,
                                   DIM, lane);
#pragma unroll
            for (int nt = 0; nt < 4; ++nt) {
                v16bf b = load_b_frag(&Bt[nt * 16][0] + kf * 32, 64, lane);
                acc[0][nt] = wmma_bf16(a0, b, acc[0][nt]);
                acc[1][nt] = wmma_bf16(a1, b, acc[1][nt]);
            }
        }
        __syncthreads();
    }

    // Epilogue: C layout — lane<16: N=lane, M=vgpr; lane>=16: M=vgpr+8
    const int nl = lane & 15;
    const int mb = (lane >> 4) * 8;
#pragma unroll
    for (int mt = 0; mt < 2; ++mt) {
#pragma unroll
        for (int nt = 0; nt < 4; ++nt) {
            const int col = n0 + nt * 16 + nl;          // 0..3071
            const int sel = col >> 10;                  // 0=Q 1=K 2=V
            const int cc  = col & 1023;
            const int h   = cc >> 6;
            const int d   = cc & 63;
            bf16_t* dst = (sel == 0) ? Q : ((sel == 1) ? K : V);
            const float s = (sel == 0) ? ATTN_SCALE : 1.0f;
#pragma unroll
            for (int i = 0; i < 8; ++i) {
                const int row = m0 + wave * 32 + mt * 16 + mb + i;  // 0..4095
                const int bb  = row >> 11;                          // batch
                const int nr  = row & 2047;                         // seq pos
                dst[((size_t)(bb * HEADS + h) * SEQ + nr) * DHEAD + d] =
                    (bf16_t)(acc[mt][nt][i] * s);
            }
        }
    }
}

// ---------------------------------------------------------------------------
// Kernel 2: flash attention. grid = (B*H, SEQ/64), block = 128 (4 waves).
// Wave w owns Q rows [w*16, w*16+16); threads 0..63 own one output row each
// for the online-softmax bookkeeping. K blocks are DMA'd into LDS by the TDM.
// ---------------------------------------------------------------------------
__global__ void __launch_bounds__(128)
attn_kernel(const bf16_t* __restrict__ Q, const bf16_t* __restrict__ K,
            const bf16_t* __restrict__ V, bf16_t* __restrict__ O) {
    __shared__ __align__(32) float  Sf[64][64];   // S tile / PV tile (reused)
    __shared__ __align__(32) bf16_t Pb[64][64];   // exp(S - m) in bf16
    __shared__ __align__(32) bf16_t Vt[64][64];   // V transposed: [d][key]
    __shared__ __align__(32) bf16_t Kb[64][64];   // K block (TDM destination)

    const int tid  = threadIdx.x;
    const int lane = tid & 31;
    const int wave = tid >> 5;
    const int bh   = blockIdx.x;          // b*HEADS + h
    const int q0   = blockIdx.y * 64;
    const size_t base = (size_t)bh * SEQ * DHEAD;

    // Q fragments for this wave's 16 rows (persist across all KV blocks)
    v16bf qf[2];
#pragma unroll
    for (int kf = 0; kf < 2; ++kf)
        qf[kf] = load_a_frag(Q + base + (size_t)(q0 + wave * 16) * DHEAD + kf * 32,
                             DHEAD, lane);

    float m = -1e30f, l = 0.0f, corr = 0.0f;
    float o[DHEAD];
#pragma unroll
    for (int d = 0; d < DHEAD; ++d) o[d] = 0.0f;

    const int nl = lane & 15;
    const int mb = (lane >> 4) * 8;

    for (int j0 = 0; j0 < SEQ; j0 += 64) {
        __syncthreads();   // all LDS readers of the previous block are done

        // Wave 0: TDM the contiguous 64x64 bf16 K block (8 KB) into LDS.
        if (wave == 0) {
            tdm_load_contig(&Kb[0][0], K + base + (size_t)j0 * DHEAD,
                            (64 * 64 * 2) / 4);
        }

        if (j0 + 64 < SEQ)   // hint the next V block toward the caches
            __builtin_prefetch(V + base + (size_t)(j0 + 64) * DHEAD, 0, 0);

        // Stage V^T : Vt[d][key]
        {
            const int r  = tid >> 1;
            const int c0 = (tid & 1) * 32;
            const bf16_t* src = V + base + (size_t)(j0 + r) * DHEAD + c0;
#pragma unroll
            for (int c = 0; c < 32; ++c) Vt[c0 + c][r] = src[c];
        }

        if (wave == 0) __builtin_amdgcn_s_wait_tensorcnt(0);
        __syncthreads();   // Kb (TDM) and Vt visible to all waves

        // S = Q @ K^T ; Kb's [key][d] LDS layout is exactly the [N][K]
        // B-fragment layout, so fragments come straight from LDS.
        v8f s[4] = {};
#pragma unroll
        for (int kf = 0; kf < 2; ++kf) {
#pragma unroll
            for (int nt = 0; nt < 4; ++nt) {
                v16bf b = load_b_frag(&Kb[nt * 16][0] + kf * 32, 64, lane);
                s[nt] = wmma_bf16(qf[kf], b, s[nt]);
            }
        }
        // Spill S strip to LDS
#pragma unroll
        for (int nt = 0; nt < 4; ++nt)
#pragma unroll
            for (int i = 0; i < 8; ++i)
                Sf[wave * 16 + mb + i][nt * 16 + nl] = s[nt][i];
        __syncthreads();

        // Online softmax: one thread per row
        if (tid < 64) {
            float bm = -1e30f;
#pragma unroll
            for (int c = 0; c < 64; ++c) bm = fmaxf(bm, Sf[tid][c]);
            const float nm = fmaxf(m, bm);
            corr = __expf(m - nm);
            float sum = 0.0f;
#pragma unroll
            for (int c = 0; c < 64; ++c) {
                const float p = __expf(Sf[tid][c] - nm);
                Pb[tid][c] = (bf16_t)p;
                sum += p;
            }
            l = l * corr + sum;
            m = nm;
        }
        __syncthreads();

        // PV = P @ V : A from Pb (row-major, key-contiguous), B from Vt[d][key]
        v8f pv[4] = {};
#pragma unroll
        for (int kf = 0; kf < 2; ++kf) {
            v16bf a = load_a_frag(&Pb[wave * 16][0] + kf * 32, 64, lane);
#pragma unroll
            for (int nt = 0; nt < 4; ++nt) {
                v16bf b = load_b_frag(&Vt[nt * 16][0] + kf * 32, 64, lane);
                pv[nt] = wmma_bf16(a, b, pv[nt]);
            }
        }
#pragma unroll
        for (int nt = 0; nt < 4; ++nt)
#pragma unroll
            for (int i = 0; i < 8; ++i)
                Sf[wave * 16 + mb + i][nt * 16 + nl] = pv[nt][i];
        __syncthreads();

        // Rescale + accumulate output rows
        if (tid < 64) {
#pragma unroll
            for (int d = 0; d < DHEAD; ++d)
                o[d] = o[d] * corr + Sf[tid][d];
        }
    }

    // Final normalize and write O in [b][n][h*DHEAD + d] bf16 layout
    if (tid < 64) {
        const int row = q0 + tid;
        const int b   = bh >> 4;
        const int h   = bh & 15;
        bf16_t* dst = O + ((size_t)b * SEQ + row) * INNER + h * DHEAD;
        const float inv = 1.0f / l;
#pragma unroll
        for (int d = 0; d < DHEAD; ++d) dst[d] = (bf16_t)(o[d] * inv);
    }
}

// ---------------------------------------------------------------------------
// Kernel 3: output GEMM  out[4096,1024] = Ob[4096,1024] @ Wob[1024,1024] + b
// Same 128x64 tile / 32x64-per-wave register blocking as the QKV GEMM.
// ---------------------------------------------------------------------------
__global__ void __launch_bounds__(128)
out_gemm_kernel(const bf16_t* __restrict__ A, const bf16_t* __restrict__ Wb,
                const float* __restrict__ bias, float* __restrict__ out) {
    __shared__ __align__(32) bf16_t Bt[64][64];   // W tile, [N][K]

    const int tid  = threadIdx.x;
    const int lane = tid & 31;
    const int wave = tid >> 5;
    const int n0   = blockIdx.x * 64;
    const int m0   = blockIdx.y * 128;

    v8f acc[2][4] = {};

    for (int k0 = 0; k0 < INNER; k0 += 64) {
        {
            const int k  = tid >> 1;
            const int c0 = (tid & 1) * 32;
            const bf16_t* src = Wb + (size_t)(k0 + k) * DIM + n0 + c0;
#pragma unroll
            for (int c = 0; c < 32; ++c) Bt[c0 + c][k] = src[c];
        }
        __syncthreads();
#pragma unroll
        for (int kf = 0; kf < 2; ++kf) {
            v16bf a0 = load_a_frag(A + (size_t)(m0 + wave * 32) * INNER + k0 + kf * 32,
                                   INNER, lane);
            v16bf a1 = load_a_frag(A + (size_t)(m0 + wave * 32 + 16) * INNER + k0 + kf * 32,
                                   INNER, lane);
#pragma unroll
            for (int nt = 0; nt < 4; ++nt) {
                v16bf b = load_b_frag(&Bt[nt * 16][0] + kf * 32, 64, lane);
                acc[0][nt] = wmma_bf16(a0, b, acc[0][nt]);
                acc[1][nt] = wmma_bf16(a1, b, acc[1][nt]);
            }
        }
        __syncthreads();
    }

    const int nl = lane & 15;
    const int mb = (lane >> 4) * 8;
#pragma unroll
    for (int mt = 0; mt < 2; ++mt) {
#pragma unroll
        for (int nt = 0; nt < 4; ++nt) {
            const int col = n0 + nt * 16 + nl;
            const float bv = bias[col];
#pragma unroll
            for (int i = 0; i < 8; ++i) {
                const int row = m0 + wave * 32 + mt * 16 + mb + i;
                out[(size_t)row * DIM + col] = acc[mt][nt][i] + bv;
            }
        }
    }
}

// ---------------------------------------------------------------------------
// Host entry
// ---------------------------------------------------------------------------
extern "C" void kernel_launch(void* const* d_in, const int* in_sizes, int n_in,
                              void* d_out, int out_size, void* d_ws, size_t ws_size,
                              hipStream_t stream) {
    const float* x     = (const float*)d_in[0];   // [2, 2048, 1024]
    const float* w_qkv = (const float*)d_in[1];   // [1024, 3072]
    const float* w_out = (const float*)d_in[2];   // [1024, 1024]
    const float* b_out = (const float*)d_in[3];   // [1024]
    float* out = (float*)d_out;                   // [2, 2048, 1024]

    char* ws = (char*)d_ws;
    size_t off = 0;
    bf16_t* Xb   = (bf16_t*)(ws + off); off += (size_t)BATCH * SEQ * DIM * 2;      // 8 MB
    bf16_t* Wqb  = (bf16_t*)(ws + off); off += (size_t)DIM * NQKV * 2;             // 6 MB
    bf16_t* Wob  = (bf16_t*)(ws + off); off += (size_t)DIM * DIM * 2;              // 2 MB
    bf16_t* Qw   = (bf16_t*)(ws + off); off += (size_t)BATCH * HEADS * SEQ * DHEAD * 2; // 8 MB
    bf16_t* Kw   = (bf16_t*)(ws + off); off += (size_t)BATCH * HEADS * SEQ * DHEAD * 2; // 8 MB
    bf16_t* Vw   = (bf16_t*)(ws + off); off += (size_t)BATCH * HEADS * SEQ * DHEAD * 2; // 8 MB
    bf16_t* Ob   = (bf16_t*)(ws + off); off += (size_t)BATCH * SEQ * INNER * 2;    // 8 MB

    cvt_bf16_kernel<<<2048, 256, 0, stream>>>(x, Xb, w_qkv, Wqb, w_out, Wob);

    qkv_gemm_kernel<<<dim3(NQKV / 64, (BATCH * SEQ) / 128), 128, 0, stream>>>(
        Xb, Wqb, Qw, Kw, Vw);

    attn_kernel<<<dim3(BATCH * HEADS, SEQ / 64), 128, 0, stream>>>(Qw, Kw, Vw, Ob);

    out_gemm_kernel<<<dim3(DIM / 64, (BATCH * SEQ) / 128), 128, 0, stream>>>(
        Ob, Wob, b_out, out);
}
